// PointSplit_8272107012334
// MI455X (gfx1250) — matrix operand order repaired and checked by reading, not compiled, not run on hardware
//
#include <hip/hip_runtime.h>
#include <hip/hip_bf16.h>

// ---------------------------------------------------------------------------
// PointNet++ MSG set-abstraction for MI455X (gfx1250, wave32, WMMA + TDM)
// ---------------------------------------------------------------------------

typedef __attribute__((ext_vector_type(16))) _Float16 v16h;
typedef __attribute__((ext_vector_type(8)))  _Float16 v8h;
typedef __attribute__((ext_vector_type(8)))  float    v8f;
typedef __attribute__((ext_vector_type(4)))  unsigned int v4u;
typedef __attribute__((ext_vector_type(8)))  int      v8i;
typedef __attribute__((ext_vector_type(4)))  int      v4i;

static constexpr int Bsz = 8;
static constexpr int Npt = 4096;
static constexpr int Sq  = 1024;
static constexpr int NQ  = Bsz * Sq;          // 8192 queries
static constexpr int INCH = 64;

// ---------------- workspace layout (bytes) ----------------
static constexpr size_t WS_IDX0 = 0;                              // 8192*16 int
static constexpr size_t WS_IDX1 = WS_IDX0 + (size_t)NQ * 16 * 4;  // 8192*32 int
static constexpr size_t WS_IDX2 = WS_IDX1 + (size_t)NQ * 32 * 4;  // 8192*64 int
static constexpr size_t WS_WF   = WS_IDX2 + (size_t)NQ * 64 * 4;  // 141312 halves
static constexpr size_t WS_BF   = WS_WF + 141312 * 2;             // 1280 floats

// ---------------------------------------------------------------------------
// Kernel 0: new_xyz = transpose(ori_xyz)  (B,S,3) -> (B,3,S)
// ---------------------------------------------------------------------------
__global__ void newxyz_kernel(const float* __restrict__ ori, float* __restrict__ out) {
    int e = blockIdx.x * blockDim.x + threadIdx.x;
    if (e >= Bsz * 3 * Sq) return;
    int b = e / (3 * Sq);
    int rem = e % (3 * Sq);
    int d = rem / Sq;
    int s = rem % Sq;
    out[e] = ori[((size_t)b * Sq + s) * 3 + d];
}

// ---------------------------------------------------------------------------
// Kernel 1: fold BN into weights -> f16 (in-ch padded to mult of 32) + f32 bias
// params per layer flattened (JAX tree / sorted keys): W, b, beta, gamma, mean, var
// ---------------------------------------------------------------------------
struct P54 { const float* q[54]; };

__global__ void prep_kernel(P54 prm, _Float16* __restrict__ Wf, float* __restrict__ Bf) {
    const int Otab[9]  = {64, 64, 128, 128, 128, 256, 128, 128, 256};
    const int Itab[9]  = {67, 64, 64,  67,  128, 128, 67,  128, 128};
    const int Ptab[9]  = {96, 64, 64,  96,  128, 128, 96,  128, 128};
    const int Woff[9]  = {0, 6144, 10240, 18432, 30720, 47104, 79872, 92160, 108544};
    const int Boff[9]  = {0, 64, 128, 256, 384, 512, 768, 896, 1024};

    const int l = blockIdx.x;       // 0..8
    const int O = Otab[l], I = Itab[l], P = Ptab[l];
    const float* W    = prm.q[l * 6 + 0];
    const float* bia  = prm.q[l * 6 + 1];
    const float* beta = prm.q[l * 6 + 2];
    const float* gam  = prm.q[l * 6 + 3];
    const float* mean = prm.q[l * 6 + 4];
    const float* var  = prm.q[l * 6 + 5];

    for (int e = threadIdx.x; e < O * P; e += blockDim.x) {
        int o = e / P, c = e % P;
        float s = gam[o] * rsqrtf(var[o] + 1e-5f);
        float w = (c < I) ? W[o * I + c] * s : 0.f;
        Wf[Woff[l] + e] = (_Float16)w;
    }
    for (int o = threadIdx.x; o < O; o += blockDim.x) {
        float s = gam[o] * rsqrtf(var[o] + 1e-5f);
        Bf[Boff[l] + o] = s * (bia[o] - mean[o]) + beta[o];
    }
}

// ---------------------------------------------------------------------------
// Kernel 2: ball query, one wave32 per query, 3 radii in one sweep.
// Ordered compaction (first K indices in ascending order) via ballot+prefix.
// ---------------------------------------------------------------------------
__global__ void ballquery_kernel(const float* __restrict__ ori,
                                 const float* __restrict__ xyz,
                                 int* __restrict__ idx0,
                                 int* __restrict__ idx1,
                                 int* __restrict__ idx2) {
    const int lane = threadIdx.x & 31;
    const int w = blockIdx.x * (blockDim.x >> 5) + (threadIdx.x >> 5); // query id
    if (w >= NQ) return;
    const int b = w >> 10;   // /Sq

    const float cx = ori[(size_t)w * 3 + 0];
    const float cy = ori[(size_t)w * 3 + 1];
    const float cz = ori[(size_t)w * 3 + 2];

    const float* xb = xyz + (size_t)b * 3 * Npt;

    int cnt0 = 0, cnt1 = 0, cnt2 = 0;
    int first0 = -1, first1 = -1, first2 = -1;

    for (int ch = 0; ch < Npt / 32; ++ch) {
        const int j = ch * 32 + lane;
        const float dx = xb[j] - cx;
        const float dy = xb[Npt + j] - cy;
        const float dz = xb[2 * Npt + j] - cz;
        const float d2 = dx * dx + dy * dy + dz * dz;
        const unsigned ltmask = (1u << lane) - 1u;

        auto process = [&](float rr2, int Kc, int* buf, int& cnt, int& first) {
            const bool p = d2 <= rr2;
            const unsigned mask = __builtin_amdgcn_ballot_w32(p);
            if (mask) {
                if (first < 0) first = ch * 32 + __builtin_ctz(mask);
                if (cnt < Kc) {
                    if (p) {
                        int pos = cnt + __builtin_popcount(mask & ltmask);
                        if (pos < Kc) buf[(size_t)w * Kc + pos] = j;
                    }
                }
                cnt += __builtin_popcount(mask);
            }
        };
        process(0.01f, 16, idx0, cnt0, first0);
        process(0.04f, 32, idx1, cnt1, first1);
        process(0.16f, 64, idx2, cnt2, first2);
        if (cnt0 >= 16 && cnt1 >= 32 && cnt2 >= 64) break;
    }

    auto pad = [&](int Kc, int* buf, int cnt, int first) {
        const int fi = (cnt > 0) ? first : (Npt - 1);
        for (int p = cnt + lane; p < Kc; p += 32) buf[(size_t)w * Kc + p] = fi;
    };
    pad(16, idx0, cnt0, first0);
    pad(32, idx1, cnt1, first1);
    pad(64, idx2, cnt2, first2);
}

// ---------------------------------------------------------------------------
// TDM: stage one weight stripe (64 out-rows x inPad halves, row-major,
// contiguous) from global into LDS.  D# per CDNA5 ISA 08_async_tensor §8.
// Issued by one wave (EXEC ignored by TDM); caller barriers afterwards.
// ---------------------------------------------------------------------------
__device__ __forceinline__ void tdm_load_stripe(const _Float16* gsrc,
                                                _Float16* wLds, int inPad) {
#if __has_builtin(__builtin_amdgcn_tensor_load_to_lds)
    const unsigned ldsOff = (unsigned)(uintptr_t)(void*)wLds;   // low 32b = LDS byte addr
    const unsigned long long ga = (unsigned long long)(uintptr_t)gsrc;
    // group0: count=1 | lds_addr | global_addr[56:0] | type=2
    v4u g0 = {1u, ldsOff, (unsigned)ga,
              (unsigned)(((ga >> 32) & 0x1FFFFFFu) | (2u << 30))};
    // group1: data_size=1(2B); tensor_dim0=inPad; tensor_dim1=64;
    //         tile_dim0=inPad; tile_dim1=64; tensor_dim0_stride=inPad
    v8i g1 = {(int)(1u << 16),
              (int)((unsigned)(inPad & 0xFFFF) << 16),
              (int)(((unsigned)(inPad >> 16) & 0xFFFF) | (64u << 16)),
              (int)((unsigned)inPad << 16),
              64,
              inPad,
              0, 0};
    v4i g2 = {0, 0, 0, 0};
    v4i g3 = {0, 0, 0, 0};
#if defined(__clang_major__) && (__clang_major__ >= 23)
    v8i g1b = {0, 0, 0, 0, 0, 0, 0, 0};
    __builtin_amdgcn_tensor_load_to_lds(g0, g1, g2, g3, g1b, 0);
#else
    __builtin_amdgcn_tensor_load_to_lds(g0, g1, g2, g3, 0);
#endif
    __builtin_amdgcn_s_wait_tensorcnt(0);
#else
    // fallback: cooperative wave copy (16B per lane per iter)
    const v8h* s = (const v8h*)gsrc;
    v8h* d = (v8h*)wLds;
    const int n = (64 * inPad) / 8;
    for (int e = threadIdx.x & 31; e < n; e += 32) d[e] = s[e];
#endif
}

// ---------------------------------------------------------------------------
// WMMA GEMM layer (all shapes compile-time):
//   [64 rows x INPAD] f16 (LDS, stride 128)  x  W'[COUT x INPAD] -> relu(.+b)
// A fragments cached in registers for the whole layer; weights staged per
// 64-column stripe into LDS by the TDM; per n-tile all B fragments are
// preloaded in one clause, then the WMMA chain issues back-to-back.
// LAST: fused max-pool -- every 16-row M-tile is inside one query
// (K multiple of 16), so reduce the 8 lane-local rows in registers first,
// then a single LDS atomicMax on the uint image (valid: values >= 0).
// Fragment layouts per CDNA5 ISA 7.12.2 (wave32).
// ---------------------------------------------------------------------------
template <int INPAD, int COUT, bool LAST, int K>
__device__ __forceinline__ void gemm_layer(const _Float16* __restrict__ actIn,
                                           const _Float16* __restrict__ Wl,
                                           const float* __restrict__ bias,
                                           _Float16* __restrict__ actOut,
                                           float* __restrict__ maxbuf,
                                           _Float16* __restrict__ wLds,
                                           int tid) {
    constexpr int kTiles = INPAD / 32;
    constexpr int nStripes = COUT / 64;

    const int lane  = tid & 31;
    const int wave  = tid >> 5;
    const int mbase = wave * 16;                 // 4 waves cover 64 rows
    const int rowA  = mbase + (lane & 15);
    const int hi    = (lane >> 4) & 1;
    const int kselA = hi * 8;                    // A: lanes>=16 hold K+8..15 / +24..31
    const int kselB = hi * 8;                    // B: lanes>=16 hold K+8..15 / +24..31
    const int col   = lane & 15;

    // ---- cache all A fragments for this layer in registers (<= 32 VGPRs)
    v16h afrag[kTiles];
#pragma unroll
    for (int kt = 0; kt < kTiles; ++kt) {
        const _Float16* ap = actIn + rowA * 128 + kt * 32 + kselA;
        v8h alo = *(const v8h*)(ap);
        v8h ahi = *(const v8h*)(ap + 16);
        afrag[kt] = __builtin_shufflevector(alo, ahi, 0, 1, 2, 3, 4, 5, 6, 7,
                                            8, 9, 10, 11, 12, 13, 14, 15);
    }

#pragma unroll
    for (int st = 0; st < nStripes; ++st) {
        if (wave == 0)
            tdm_load_stripe(Wl + (size_t)st * 64 * INPAD, wLds, INPAD);
        __syncthreads();

#pragma unroll
        for (int nt = 0; nt < 4; ++nt) {
            const int ocol = st * 64 + nt * 16 + col;
            const _Float16* wrow = wLds + (nt * 16 + col) * INPAD + kselB;

            // preload all B fragments for this n-tile in one clause
            v16h bfrag[kTiles];
#pragma unroll
            for (int kt = 0; kt < kTiles; ++kt) {
                const _Float16* bp = wrow + kt * 32;
                v8h blo = *(const v8h*)(bp);
                v8h bhi = *(const v8h*)(bp + 16);
                bfrag[kt] = __builtin_shufflevector(blo, bhi, 0, 1, 2, 3, 4, 5, 6, 7,
                                                    8, 9, 10, 11, 12, 13, 14, 15);
            }

            v8f acc = {};
#pragma unroll
            for (int kt = 0; kt < kTiles; ++kt)
                acc = __builtin_amdgcn_wmma_f32_16x16x32_f16(false, afrag[kt], false,
                                                             bfrag[kt], (short)0, acc,
                                                             false, false);

            const float bv = bias[ocol];
            if constexpr (!LAST) {
#pragma unroll
                for (int v = 0; v < 8; ++v) {
                    float val = fmaxf(acc[v] + bv, 0.f);
                    int row = mbase + v + hi * 8;
                    actOut[row * 128 + ocol] = (_Float16)val;
                }
            } else {
                // all 8 rows in this lane belong to the same query (K % 16 == 0)
                float m = acc[0];
#pragma unroll
                for (int v = 1; v < 8; ++v) m = fmaxf(m, acc[v]);
                float val = fmaxf(m + bv, 0.f);
                const int ql = mbase / K;
                atomicMax((unsigned int*)&maxbuf[ql * COUT + ocol],
                          __float_as_uint(val));
            }
        }
        __syncthreads();                         // wLds reused by next stripe
    }
}

// ---------------------------------------------------------------------------
// Kernel 3: per-branch gather + 3-layer MLP + maxpool.  128 threads = 4 waves,
// 64 (query,sample) rows per block -> QPB = 64/K queries.
// ---------------------------------------------------------------------------
template <int K, int QPB, int C1, int C2, int C3, int CH_OFF>
__global__ void mlp_kernel(const float* __restrict__ ori,
                           const float* __restrict__ xyz,
                           const float* __restrict__ points,
                           const int* __restrict__ idxbuf,
                           const _Float16* __restrict__ Wf,
                           const float* __restrict__ Bf,
                           int w1, int w2, int w3, int b1, int b2, int b3,
                           float* __restrict__ out) {
    __shared__ _Float16 bufA[64 * 128];          // 16 KB
    __shared__ _Float16 bufB[64 * 128];          // 16 KB
    __shared__ _Float16 wBuf[64 * 128];          // 16 KB weight stripe stage
    __shared__ float    maxbuf[QPB * C3];

    const int tid = threadIdx.x;
    const int g0  = blockIdx.x * QPB;            // first global query of block

    for (int e = tid; e < QPB * C3; e += 128) maxbuf[e] = 0.f;

    // ---- gather: rows = (q_local, samp); cols 0..63 feats, 64..66 rel xyz, pad 0
    {
        const int r = tid >> 1;                  // 0..63
        const int half = tid & 1;
        const int q_local = r / K;
        const int samp = r % K;
        const int g = g0 + q_local;
        const int b = g >> 10;
        const int j = idxbuf[(size_t)g * K + samp];
        const float cx = ori[(size_t)g * 3 + 0];
        const float cy = ori[(size_t)g * 3 + 1];
        const float cz = ori[(size_t)g * 3 + 2];
        const float* pb = points + (size_t)b * INCH * Npt;
        const float* xb = xyz + (size_t)b * 3 * Npt;
        const int c0 = half * 48;
        for (int c = c0; c < c0 + 48; ++c) {
            float v;
            if (c < 64)       v = pb[(size_t)c * Npt + j];
            else if (c == 64) v = xb[j] - cx;
            else if (c == 65) v = xb[Npt + j] - cy;
            else if (c == 66) v = xb[2 * Npt + j] - cz;
            else              v = 0.f;
            bufA[r * 128 + c] = (_Float16)v;
        }
    }
    __syncthreads();

    gemm_layer<96, C1, false, K>(bufA, Wf + w1, Bf + b1, bufB, nullptr, wBuf, tid);
    __syncthreads();
    gemm_layer<C1, C2, false, K>(bufB, Wf + w2, Bf + b2, bufA, nullptr, wBuf, tid);
    __syncthreads();
    gemm_layer<C2, C3, true, K>(bufA, Wf + w3, Bf + b3, nullptr, maxbuf, wBuf, tid);
    __syncthreads();

    // ---- write: out is new_points (B,640,S), after 24576-float new_xyz prefix
    for (int e = tid; e < QPB * C3; e += 128) {
        const int ql = e / C3;
        const int o  = e % C3;
        const int g  = g0 + ql;
        const int b  = g >> 10;
        const int s  = g & 1023;
        out[24576 + ((size_t)(b * 640 + CH_OFF + o)) * Sq + s] = maxbuf[e];
    }
}

// ---------------------------------------------------------------------------
extern "C" void kernel_launch(void* const* d_in, const int* in_sizes, int n_in,
                              void* d_out, int out_size, void* d_ws, size_t ws_size,
                              hipStream_t stream) {
    const float* ori    = (const float*)d_in[0];   // (B,S,3)
    const float* xyz    = (const float*)d_in[1];   // (B,3,N)
    const float* points = (const float*)d_in[2];   // (B,64,N)
    float* out = (float*)d_out;

    char* ws = (char*)d_ws;
    int*       idx0 = (int*)(ws + WS_IDX0);
    int*       idx1 = (int*)(ws + WS_IDX1);
    int*       idx2 = (int*)(ws + WS_IDX2);
    _Float16*  Wf   = (_Float16*)(ws + WS_WF);
    float*     Bf   = (float*)(ws + WS_BF);

    P54 prm;
    for (int i = 0; i < 54; ++i) prm.q[i] = (const float*)d_in[3 + i];

    newxyz_kernel<<<dim3((Bsz * 3 * Sq + 255) / 256), dim3(256), 0, stream>>>(ori, out);
    prep_kernel<<<dim3(9), dim3(256), 0, stream>>>(prm, Wf, Bf);
    ballquery_kernel<<<dim3(NQ / 8), dim3(256), 0, stream>>>(ori, xyz, idx0, idx1, idx2);

    // branch 0: K=16, 67->64->64->128, ch offset 0
    mlp_kernel<16, 4, 64, 64, 128, 0>
        <<<dim3(NQ / 4), dim3(128), 0, stream>>>(ori, xyz, points, idx0, Wf, Bf,
                                                 0, 6144, 10240, 0, 64, 128, out);
    // branch 1: K=32, 67->128->128->256, ch offset 128
    mlp_kernel<32, 2, 128, 128, 256, 128>
        <<<dim3(NQ / 2), dim3(128), 0, stream>>>(ori, xyz, points, idx1, Wf, Bf,
                                                 18432, 30720, 47104, 256, 384, 512, out);
    // branch 2: K=64, 67->128->128->256, ch offset 384
    mlp_kernel<64, 1, 128, 128, 256, 384>
        <<<dim3(NQ / 1), dim3(128), 0, stream>>>(ori, xyz, points, idx2, Wf, Bf,
                                                 79872, 92160, 108544, 768, 896, 1024, out);
}
